// SSMnet_30614526886200
// MI455X (gfx1250) — matrix operand order repaired and checked by reading, not compiled, not run on hardware
//
#include <hip/hip_runtime.h>
#include <hip/hip_bf16.h>

typedef __attribute__((ext_vector_type(16))) _Float16 v16h;
typedef __attribute__((ext_vector_type(8)))  float    v8f;

#define BN_EPS 1e-5f

// ---------------- f32 -> f16 convert ----------------
__global__ void k_cvt_h(const float* __restrict__ src, _Float16* __restrict__ dst, int n) {
    int i = blockIdx.x * blockDim.x + threadIdx.x;
    if (i < n) dst[i] = (_Float16)src[i];
}

// ---------------- conv1: direct, fused bias/relu/bn/maxpool2x2 ----------------
// x: [512][72][64], w: [16][1][6][4]  ->  y: [512][16][36][32] f16
__global__ void k_conv1(const float* __restrict__ x, const float* __restrict__ w,
                        const float* __restrict__ b, const float* __restrict__ g,
                        const float* __restrict__ bt, const float* __restrict__ mu,
                        const float* __restrict__ var, _Float16* __restrict__ y)
{
    int o = blockIdx.x * blockDim.x + threadIdx.x;
    const int total = 512 * 16 * 36 * 32;
    if (o >= total) return;
    int pw  = o & 31;
    int ph  = (o >> 5) % 36;
    int c   = (o / (36 * 32)) & 15;
    int img = o / (16 * 36 * 32);
    const float* xi = x + (size_t)img * 72 * 64;
    const float* wc = w + c * 24;
    float scale = g[c] * rsqrtf(var[c] + BN_EPS);
    float best = -1e30f;
    for (int dy = 0; dy < 2; ++dy)
    for (int dx = 0; dx < 2; ++dx) {
        int oh = 2 * ph + dy, ow = 2 * pw + dx;
        float acc = b[c];
        #pragma unroll
        for (int kh = 0; kh < 6; ++kh) {
            int ih = oh + kh - 2;
            if (ih < 0 || ih >= 72) continue;
            #pragma unroll
            for (int kw = 0; kw < 4; ++kw) {
                int iw = ow + kw - 1;
                if (iw < 0 || iw >= 64) continue;
                acc += xi[ih * 64 + iw] * wc[kh * 4 + kw];
            }
        }
        acc = fmaxf(acc, 0.f);
        acc = (acc - mu[c]) * scale + bt[c];
        best = fmaxf(best, acc);
    }
    y[o] = (_Float16)best;
}

// ---------------- implicit-GEMM conv (6x4, pad (2,3)/(1,2)) via WMMA ----------------
// in16: [nimg][CIN][IH][IW], w16: [COUT][CIN*24] (OIHW flat), out16: [nimg][COUT][IH*IW]
// Each wave owns one 16-column (nt) tile and carries MT accumulators so the
// expensive im2col B-gather from LDS is done once per k-chunk for all M-tiles.
template<int CIN, int IH, int IW, int COUT>
__global__ __launch_bounds__(256)
void k_conv_wmma(const _Float16* __restrict__ in16, const _Float16* __restrict__ w16,
                 const float* __restrict__ bias, const float* __restrict__ g,
                 const float* __restrict__ bt, const float* __restrict__ mu,
                 const float* __restrict__ var, _Float16* __restrict__ out16)
{
    constexpr int OW = IW;
    constexpr int PH = IH + 5, PW = IW + 3;
    constexpr int K  = CIN * 24;
    constexpr int N  = IH * IW;
    constexpr int NT = N / 16;
    constexpr int MT = COUT / 16;
    __shared__ _Float16 simg[CIN * PH * PW];

    const int img  = blockIdx.x;
    const int tid  = threadIdx.x;
    const int lane = tid & 31;
    const int wave = tid >> 5;

    // pull the (L2-hot) weight panel toward this WGP
    for (int p = tid * 64; p < COUT * K * 2; p += 256 * 64)
        __builtin_prefetch((const char*)w16 + p, 0, 1);

    // stage zero-padded image in LDS
    for (int i = tid; i < CIN * PH * PW; i += 256) simg[i] = (_Float16)0.f;
    __syncthreads();
    const _Float16* src = in16 + (size_t)img * CIN * IH * IW;
    for (int i = tid; i < CIN * IH * IW; i += 256) {
        int w_ = i % IW;
        int h_ = (i / IW) % IH;
        int c_ = i / (IH * IW);
        simg[c_ * PH * PW + (h_ + 2) * PW + (w_ + 1)] = src[i];
    }
    __syncthreads();

    const int mrow  = lane & 15;
    const int kselA = (lane < 16) ? 0 : 16;   // A: elems 0-15 -> K+0..15 / 16..31
    const int kselB = (lane < 16) ? 0 : 16;   // B: elems 0-15 -> K+0..15 / 16..31
    const int ncol  = lane & 15;
    const int hi8   = (lane < 16) ? 0 : 8;

    for (int nt = wave; nt < NT; nt += 8) {
        const int n  = nt * 16 + ncol;
        const int oh = n / OW, ow = n % OW;
        v8f acc[MT];
        const v8f z = {};
        #pragma unroll
        for (int mt = 0; mt < MT; ++mt) acc[mt] = z;

        for (int kc = 0; kc < K / 32; ++kc) {
            // B fragment: gathered once, reused by all MT wmma ops
            v16h bfr;
            const int kb = kc * 32 + kselB;
            #pragma unroll
            for (int e = 0; e < 16; ++e) {
                int k  = kb + e;
                int c_ = k / 24;
                int r_ = k - c_ * 24;
                bfr[e] = simg[c_ * PH * PW + (oh + (r_ >> 2)) * PW + (ow + (r_ & 3))];
            }
            #pragma unroll
            for (int mt = 0; mt < MT; ++mt) {
                v16h a;
                const _Float16* wr = w16 + (size_t)(mt * 16 + mrow) * K + kc * 32 + kselA;
                #pragma unroll
                for (int e = 0; e < 8; ++e) a[e]     = wr[e];
                #pragma unroll
                for (int e = 0; e < 8; ++e) a[8 + e] = wr[8 + e];
                acc[mt] = __builtin_amdgcn_wmma_f32_16x16x32_f16(false, a, false, bfr,
                                                                 (short)0, acc[mt], false, false);
            }
        }
        #pragma unroll
        for (int mt = 0; mt < MT; ++mt) {
            #pragma unroll
            for (int r = 0; r < 8; ++r) {
                int mo = mt * 16 + r + hi8;
                float val = acc[mt][r] + bias[mo];
                val = fmaxf(val, 0.f);
                val = (val - mu[mo]) * (g[mo] * rsqrtf(var[mo] + BN_EPS)) + bt[mo];
                out16[((size_t)img * COUT + mo) * N + nt * 16 + ncol] = (_Float16)val;
            }
        }
    }
}

// ---------------- generic maxpool on f16 ----------------
// in: [B*C][H][W] -> out: [B*C][H/ph][W/pw]
__global__ void k_pool(const _Float16* __restrict__ in, _Float16* __restrict__ out,
                       int H, int W, int ph, int pw, int total)
{
    int o = blockIdx.x * blockDim.x + threadIdx.x;
    if (o >= total) return;
    int OW = W / pw, OH = H / ph;
    int ow = o % OW;
    int oh = (o / OW) % OH;
    int cc = o / (OW * OH);
    const _Float16* base = in + ((size_t)cc * H + oh * ph) * W + (size_t)ow * pw;
    float best = -1e30f;
    for (int y = 0; y < ph; ++y)
        for (int x = 0; x < pw; ++x)
            best = fmaxf(best, (float)base[y * W + x]);
    out[o] = (_Float16)best;
}

// ---------------- FC layer via WMMA: Y = bn(relu(X @ W^T + b)) ----------------
// X: [M][K] f16, W: [N][K] f16 (fc weight layout). grid.x*8 waves == (M/16)*(N/16)
__global__ __launch_bounds__(256)
void k_fc_wmma(const _Float16* __restrict__ X, const _Float16* __restrict__ W,
               const float* __restrict__ bias, const float* __restrict__ g,
               const float* __restrict__ bt, const float* __restrict__ mu,
               const float* __restrict__ var, float* __restrict__ Yf,
               _Float16* __restrict__ Yh, int N, int K)
{
    const int lane = threadIdx.x & 31;
    const int wave = threadIdx.x >> 5;
    const int NT = N >> 4;
    const int t  = blockIdx.x * 8 + wave;
    const int mt = t / NT, nt = t % NT;
    const int mrow  = lane & 15;
    const int kselA = (lane < 16) ? 0 : 16;
    const int kselB = (lane < 16) ? 0 : 16;
    const int ncol  = lane & 15;
    const _Float16* xr = X + (size_t)(mt * 16 + mrow) * K;
    const _Float16* wc = W + (size_t)(nt * 16 + ncol) * K;
    v8f acc = {};
    for (int kc = 0; kc < K / 32; ++kc) {
        v16h a, b;
        const _Float16* xa = xr + kc * 32 + kselA;
        #pragma unroll
        for (int e = 0; e < 8; ++e) a[e]     = xa[e];
        #pragma unroll
        for (int e = 0; e < 8; ++e) a[8 + e] = xa[8 + e];
        const _Float16* wb = wc + kc * 32 + kselB;
        #pragma unroll
        for (int e = 0; e < 16; ++e) b[e] = wb[e];
        acc = __builtin_amdgcn_wmma_f32_16x16x32_f16(false, a, false, b,
                                                     (short)0, acc, false, false);
    }
    #pragma unroll
    for (int r = 0; r < 8; ++r) {
        int mm = mt * 16 + r + ((lane < 16) ? 0 : 8);
        int nn = nt * 16 + ncol;
        float val = acc[r] + bias[nn];
        val = fmaxf(val, 0.f);
        val = (val - mu[nn]) * (g[nn] * rsqrtf(var[nn] + BN_EPS)) + bt[nn];
        Yf[(size_t)mm * N + nn] = val;
        Yh[(size_t)mm * N + nn] = (_Float16)val;
    }
}

// ---------------- L2 normalize rows of [512][128] ----------------
__global__ void k_normalize(const float* __restrict__ X, float* __restrict__ E,
                            _Float16* __restrict__ Eh, float* __restrict__ sq)
{
    __shared__ float red[128];
    int row = blockIdx.x, i = threadIdx.x;
    float xv = X[row * 128 + i];
    red[i] = xv * xv;
    __syncthreads();
    for (int s = 64; s > 0; s >>= 1) {
        if (i < s) red[i] += red[i + s];
        __syncthreads();
    }
    float ss    = red[0];
    float denom = fmaxf(sqrtf(ss), 1e-12f);
    float e     = xv / denom;
    E[row * 128 + i]  = e;
    Eh[row * 128 + i] = (_Float16)e;
    if (i == 0) sq[row] = ss / (denom * denom);
}

// ---------------- Gram + distance + global max (WMMA) ----------------
__global__ __launch_bounds__(256)
void k_gram_dist(const _Float16* __restrict__ E, const float* __restrict__ sq,
                 float* __restrict__ dmat, unsigned int* __restrict__ maxbits)
{
    __shared__ float red[256];
    const int lane = threadIdx.x & 31;
    const int wave = threadIdx.x >> 5;
    const int t  = blockIdx.x * 8 + wave;       // 1024 tiles (32x32)
    const int mt = t >> 5, nt = t & 31;
    const int mrow  = lane & 15;
    const int kselA = (lane < 16) ? 0 : 16;
    const int kselB = (lane < 16) ? 0 : 16;
    const int ncol  = lane & 15;
    const _Float16* ar = E + (size_t)(mt * 16 + mrow) * 128;
    const _Float16* br = E + (size_t)(nt * 16 + ncol) * 128;
    v8f acc = {};
    for (int kc = 0; kc < 4; ++kc) {
        v16h a, b;
        #pragma unroll
        for (int e = 0; e < 8; ++e) a[e]     = ar[kc * 32 + kselA + e];
        #pragma unroll
        for (int e = 0; e < 8; ++e) a[8 + e] = ar[kc * 32 + kselA + 8 + e];
        #pragma unroll
        for (int e = 0; e < 16; ++e) b[e] = br[kc * 32 + kselB + e];
        acc = __builtin_amdgcn_wmma_f32_16x16x32_f16(false, a, false, b,
                                                     (short)0, acc, false, false);
    }
    float lmax = 0.f;
    #pragma unroll
    for (int r = 0; r < 8; ++r) {
        int mm = mt * 16 + r + ((lane < 16) ? 0 : 8);
        int nn = nt * 16 + ncol;
        float d2 = sq[mm] + sq[nn] - 2.f * acc[r];
        float d  = sqrtf(fmaxf(d2, 1e-12f));
        d = (d2 > 0.f) ? d : 0.f;
        dmat[(size_t)mm * 512 + nn] = d;
        lmax = fmaxf(lmax, d);
    }
    red[threadIdx.x] = lmax;
    __syncthreads();
    for (int s = 128; s > 0; s >>= 1) {
        if (threadIdx.x < s) red[threadIdx.x] = fmaxf(red[threadIdx.x], red[threadIdx.x + s]);
        __syncthreads();
    }
    if (threadIdx.x == 0) atomicMax(maxbits, __float_as_uint(red[0]));
}

__global__ void k_init_max(unsigned int* maxbits) {
    if (threadIdx.x == 0 && blockIdx.x == 0) *maxbits = 0u;
}

__global__ void k_scale(const float* __restrict__ dmat,
                        const unsigned int* __restrict__ maxbits, float* __restrict__ out)
{
    int i = blockIdx.x * blockDim.x + threadIdx.x;
    if (i < 512 * 512) {
        float mx = __uint_as_float(*maxbits);
        out[i] = dmat[i] / mx;
    }
}

// ---------------- host ----------------
static inline size_t aln(size_t x) { return (x + 255) & ~(size_t)255; }

extern "C" void kernel_launch(void* const* d_in, const int* in_sizes, int n_in,
                              void* d_out, int out_size, void* d_ws, size_t ws_size,
                              hipStream_t stream)
{
    (void)in_sizes; (void)n_in; (void)out_size; (void)ws_size;
    const float* x     = (const float*)d_in[0];
    const float* c1w   = (const float*)d_in[1];
    const float* c1b   = (const float*)d_in[2];
    const float* b1g   = (const float*)d_in[3];
    const float* b1b   = (const float*)d_in[4];
    const float* b1m   = (const float*)d_in[5];
    const float* b1v   = (const float*)d_in[6];
    const float* c2w   = (const float*)d_in[7];
    const float* c2b   = (const float*)d_in[8];
    const float* b2g   = (const float*)d_in[9];
    const float* b2b   = (const float*)d_in[10];
    const float* b2m   = (const float*)d_in[11];
    const float* b2v   = (const float*)d_in[12];
    const float* c3w   = (const float*)d_in[13];
    const float* c3b   = (const float*)d_in[14];
    const float* b3g   = (const float*)d_in[15];
    const float* b3b   = (const float*)d_in[16];
    const float* b3m   = (const float*)d_in[17];
    const float* b3v   = (const float*)d_in[18];
    const float* f1w   = (const float*)d_in[19];
    const float* f1b   = (const float*)d_in[20];
    const float* n1g   = (const float*)d_in[21];
    const float* n1b   = (const float*)d_in[22];
    const float* n1m   = (const float*)d_in[23];
    const float* n1v   = (const float*)d_in[24];
    const float* f2w   = (const float*)d_in[25];
    const float* f2b   = (const float*)d_in[26];
    const float* n2g   = (const float*)d_in[27];
    const float* n2b   = (const float*)d_in[28];
    const float* n2m   = (const float*)d_in[29];
    const float* n2v   = (const float*)d_in[30];
    const float* f3w   = (const float*)d_in[31];
    const float* f3b   = (const float*)d_in[32];
    const float* n3g   = (const float*)d_in[33];
    const float* n3b   = (const float*)d_in[34];
    const float* n3m   = (const float*)d_in[35];
    const float* n3v   = (const float*)d_in[36];
    float* out = (float*)d_out;

    // ---- carve workspace ----
    char* ws = (char*)d_ws;
    size_t off = 0;
    auto grab = [&](size_t bytes) { char* p = ws + off; off += aln(bytes); return p; };
    _Float16* W2h   = (_Float16*)grab(12288 * 2);            // [32][384]
    _Float16* W3h   = (_Float16*)grab(49152 * 2);            // [64][768]
    _Float16* WF1h  = (_Float16*)grab(786432 * 2);           // [512][1536]
    _Float16* WF2h  = (_Float16*)grab(131072 * 2);           // [256][512]
    _Float16* WF3h  = (_Float16*)grab(32768 * 2);            // [128][256]
    _Float16* Y1h   = (_Float16*)grab((size_t)512 * 16 * 1152 * 2);  // conv1 pooled
    _Float16* CBUF  = (_Float16*)grab((size_t)512 * 32 * 1152 * 2);  // conv2 out, reused for conv3 out
    _Float16* Y2h   = (_Float16*)grab((size_t)512 * 32 * 288 * 2);   // conv2 pooled
    _Float16* XFLAT = (_Float16*)grab((size_t)512 * 1536 * 2);       // conv3 pooled == flattened
    float*    FC1f  = (float*)grab((size_t)512 * 512 * 4);
    _Float16* FC1h  = (_Float16*)grab((size_t)512 * 512 * 2);
    float*    FC2f  = (float*)grab((size_t)512 * 256 * 4);
    _Float16* FC2h  = (_Float16*)grab((size_t)512 * 256 * 2);
    float*    FC3f  = (float*)grab((size_t)512 * 128 * 4);
    _Float16* FC3h  = (_Float16*)grab((size_t)512 * 128 * 2);
    float*    E32   = (float*)grab((size_t)512 * 128 * 4);
    _Float16* E16   = (_Float16*)grab((size_t)512 * 128 * 2);
    float*    SQ    = (float*)grab(512 * 4);
    float*    DMAT  = (float*)grab((size_t)512 * 512 * 4);
    unsigned int* MAXB = (unsigned int*)grab(256);

    // ---- weight conversions to f16 ----
    k_cvt_h<<<48,   256, 0, stream>>>(c2w, W2h,  12288);
    k_cvt_h<<<192,  256, 0, stream>>>(c3w, W3h,  49152);
    k_cvt_h<<<3072, 256, 0, stream>>>(f1w, WF1h, 786432);
    k_cvt_h<<<512,  256, 0, stream>>>(f2w, WF2h, 131072);
    k_cvt_h<<<128,  256, 0, stream>>>(f3w, WF3h, 32768);

    // ---- conv1 (direct + pool2x2) ----
    k_conv1<<<36864, 256, 0, stream>>>(x, c1w, c1b, b1g, b1b, b1m, b1v, Y1h);

    // ---- conv2 (WMMA implicit GEMM) + pool2x2 ----
    k_conv_wmma<16, 36, 32, 32><<<512, 256, 0, stream>>>(Y1h, W2h, c2b, b2g, b2b, b2m, b2v, CBUF);
    k_pool<<<18432, 256, 0, stream>>>(CBUF, Y2h, 36, 32, 2, 2, 512 * 32 * 18 * 16);

    // ---- conv3 (WMMA implicit GEMM) + pool3x4 -> flat [512][1536] ----
    k_conv_wmma<32, 18, 16, 64><<<512, 256, 0, stream>>>(Y2h, W3h, c3b, b3g, b3b, b3m, b3v, CBUF);
    k_pool<<<3072, 256, 0, stream>>>(CBUF, XFLAT, 18, 16, 3, 4, 512 * 64 * 6 * 4);

    // ---- FC layers (WMMA) ----
    k_fc_wmma<<<128, 256, 0, stream>>>(XFLAT, WF1h, f1b, n1g, n1b, n1m, n1v, FC1f, FC1h, 512, 1536);
    k_fc_wmma<<<64,  256, 0, stream>>>(FC1h,  WF2h, f2b, n2g, n2b, n2m, n2v, FC2f, FC2h, 256, 512);
    k_fc_wmma<<<32,  256, 0, stream>>>(FC2h,  WF3h, f3b, n3g, n3b, n3m, n3v, FC3f, FC3h, 128, 256);

    // ---- normalize + pairwise distances + max + scale ----
    k_normalize<<<512, 128, 0, stream>>>(FC3f, E32, E16, SQ);
    k_init_max<<<1, 32, 0, stream>>>(MAXB);
    k_gram_dist<<<128, 256, 0, stream>>>(E16, SQ, DMAT, MAXB);
    k_scale<<<1024, 256, 0, stream>>>(DMAT, MAXB, out);
}
